// Log_GraphConv4d_73624329388051
// MI455X (gfx1250) — matrix operand order repaired and checked by reading, not compiled, not run on hardware
//
#include <hip/hip_runtime.h>
#include <hip/hip_bf16.h>
#include <math.h>

typedef __bf16 bf16_t;
typedef __attribute__((ext_vector_type(16))) __bf16 v16bf;
typedef __attribute__((ext_vector_type(8)))  float  v8f;
typedef __attribute__((ext_vector_type(4)))  unsigned int v4u;

#define COUT 384
#define CIN  384
#define KDIM 768
#define HH   56
#define WW   56
#define HW   3136
#define NB   16

// ---------------- Pass 0: fp32 -> bf16 weight conversion ----------------
__global__ __launch_bounds__(256) void cvt_w_kernel(const float* __restrict__ w,
                                                    bf16_t* __restrict__ wb, int n) {
    int i = blockIdx.x * 256 + threadIdx.x;
    if (i < n) wb[i] = (bf16_t)w[i];
}

// ---------------- Pass 1: log-shift-max, emit [x ; x_j] as bf16 ----------------
// One workgroup per (b,c) 56x56 plane; plane staged in LDS.
__global__ __launch_bounds__(256) void shiftmax_kernel(const float* __restrict__ x,
                                                       bf16_t* __restrict__ xcb) {
    __shared__ float pl[HW];
    const int bc = blockIdx.x;              // 0 .. 16*384-1
    const int b = bc / CIN;
    const int c = bc % CIN;
    const float* xp = x + (size_t)bc * HW;

    for (int p = threadIdx.x; p < HW; p += 256) pl[p] = xp[p];
    __syncthreads();

    bf16_t* dx  = xcb + ((size_t)b * KDIM + c) * HW;          // channels 0..383 : x
    bf16_t* dxj = xcb + ((size_t)b * KDIM + CIN + c) * HW;    // channels 384..767 : x_j

    const int sh[5] = {1, 3, 7, 15, 31};                      // 2^i - 1, i=1..5 (bitlen(56)=6)
    for (int p = threadIdx.x; p < HW; p += 256) {
        const int h = p / WW, w = p % WW;
        const float xv = pl[p];
        float m = 0.0f;                                        // x_j starts at zeros
#pragma unroll
        for (int i = 0; i < 5; ++i) {
            const int s = sh[i];
            int hm = h - s; if (hm < 0)   hm += HH;            // roll(+s): x[h]-x[h-s]
            int hp = h + s; if (hp >= HH) hp -= HH;            // roll(-s): x[h]-x[h+s]
            m = fmaxf(m, xv - pl[hm * WW + w]);
            m = fmaxf(m, xv - pl[hp * WW + w]);
            int wm = w - s; if (wm < 0)   wm += WW;
            int wp = w + s; if (wp >= WW) wp -= WW;
            m = fmaxf(m, xv - pl[h * WW + wm]);
            m = fmaxf(m, xv - pl[h * WW + wp]);
        }
        dx[p]  = (bf16_t)xv;
        dxj[p] = (bf16_t)m;
    }
}

// ---------------- Pass 2: bf16 WMMA GEMM + bias + BN + exact GELU ----------------
// Per workgroup: 128(M) x 64(N) output tile, 8 waves, each wave 16x64.
// A = wb [384,768] bf16 row-major (M,K); B = xcb[b] [768,3136] bf16 (K,N).
#define LDSB_STRIDE 40   // [n][k] with pad: 40*2B = 80B row stride (16B aligned)

__global__ __launch_bounds__(256) void gemm_bn_gelu_kernel(
    const bf16_t* __restrict__ wb, const bf16_t* __restrict__ xcb,
    const float* __restrict__ cb,  const float* __restrict__ bnS,
    const float* __restrict__ bnB, const float* __restrict__ bnM,
    const float* __restrict__ bnV, float* __restrict__ out) {

    __shared__ bf16_t ldsB[64 * LDSB_STRIDE];

    const int tid   = threadIdx.x;
    const int wave  = tid >> 5;
    const int lane  = tid & 31;
    const int lhalf = lane >> 4;       // 0: lanes 0-15, 1: lanes 16-31
    const int lcol  = lane & 15;
    const int nBase = blockIdx.x * 64; // 3136 = 49*64
    const int mBase = blockIdx.y * 128;// 384  = 3*128
    const int b     = blockIdx.z;

    const bf16_t* xb = xcb + (size_t)b * KDIM * HW;

    v8f acc[4] = {};                   // four 16x16 f32 accumulators (N sub-tiles)

    const int mRow = mBase + wave * 16 + lcol;              // A-matrix row for this lane
    const bf16_t* wrow = wb + (size_t)mRow * KDIM;
    const int aoff = lhalf ? 8 : 0;                         // A layout: lanes<16 K0-7/16-23, else K8-15/24-31

    const int kb    = tid >> 3;        // 0..31 : K row of B tile to stage
    const int chunk = tid & 7;         // 0..7  : 8-element N chunk

    for (int k0 = 0; k0 < KDIM; k0 += 32) {
        // Cooperative stage of B tile [32 K x 64 N], transposed into LDS [n][k]
        union { v4u u; bf16_t h[8]; } ld;
        ld.u = *(const v4u*)(xb + (size_t)(k0 + kb) * HW + nBase + chunk * 8);
#pragma unroll
        for (int e = 0; e < 8; ++e)
            ldsB[(chunk * 8 + e) * LDSB_STRIDE + kb] = ld.h[e];

        if (k0 + 32 < KDIM)            // global_prefetch_b8 for next K-tile
            __builtin_prefetch(xb + (size_t)(k0 + 32 + kb) * HW + nBase + chunk * 8, 0, 1);
        __syncthreads();

        // A fragment: 16x32 bf16, two 16B chunks per lane
        union { v16bf v; v4u u[2]; } afrag;
        afrag.u[0] = *(const v4u*)(wrow + k0 + aoff);
        afrag.u[1] = *(const v4u*)(wrow + k0 + 16 + aoff);

#pragma unroll
        for (int j = 0; j < 4; ++j) {
            // B fragment: 32x16 bf16; lanes 0-15 hold K0-7/16-23, lanes 16-31 hold K8-15/24-31
            union { v16bf v; v4u u[2]; } bfrag;
            const bf16_t* bp = &ldsB[(j * 16 + lcol) * LDSB_STRIDE + (lhalf ? 8 : 0)];
            bfrag.u[0] = *(const v4u*)(bp);
            bfrag.u[1] = *(const v4u*)(bp + 16);
            acc[j] = __builtin_amdgcn_wmma_f32_16x16x32_bf16(
                false, afrag.v, false, bfrag.v, (short)0, acc[j], false, false);
        }
        __syncthreads();
    }

    // Epilogue: bias + BN(eval) + exact GELU, C/D layout: VGPR r -> M = r + 8*lhalf, N = lcol
#pragma unroll
    for (int j = 0; j < 4; ++j) {
        const int n = nBase + j * 16 + lcol;
#pragma unroll
        for (int r = 0; r < 8; ++r) {
            const int m = mBase + wave * 16 + lhalf * 8 + r;
            float y = acc[j][r] + cb[m];
            const float inv = rsqrtf(bnV[m] + 1e-5f);
            y = (y - bnM[m]) * (inv * bnS[m]) + bnB[m];
            const float g = 0.5f * y * (1.0f + erff(y * 0.70710678118654752f));
            out[((size_t)b * COUT + m) * HW + n] = g;
        }
    }
}

// ---------------- Launch ----------------
extern "C" void kernel_launch(void* const* d_in, const int* in_sizes, int n_in,
                              void* d_out, int out_size, void* d_ws, size_t ws_size,
                              hipStream_t stream) {
    const float* x        = (const float*)d_in[0];
    const float* conv_w   = (const float*)d_in[1];
    const float* conv_b   = (const float*)d_in[2];
    const float* bn_scale = (const float*)d_in[3];
    const float* bn_bias  = (const float*)d_in[4];
    const float* bn_mean  = (const float*)d_in[5];
    const float* bn_var   = (const float*)d_in[6];
    float* out = (float*)d_out;

    // Workspace: [0, 1MB) bf16 weights; [1MB, 1MB+77MB) bf16 xc = [x ; x_j]
    bf16_t* wb  = (bf16_t*)d_ws;
    bf16_t* xcb = (bf16_t*)((char*)d_ws + (1u << 20));

    const int nW = COUT * KDIM; // 294912
    cvt_w_kernel<<<(nW + 255) / 256, 256, 0, stream>>>(conv_w, wb, nW);

    shiftmax_kernel<<<NB * CIN, 256, 0, stream>>>(x, xcb);

    dim3 grid(HW / 64, COUT / 128, NB); // (49, 3, 16)
    gemm_bn_gelu_kernel<<<grid, 256, 0, stream>>>(wb, xcb, conv_b, bn_scale, bn_bias,
                                                  bn_mean, bn_var, out);
}
